// Head_30322469110484
// MI455X (gfx1250) — matrix operand order repaired and checked by reading, not compiled, not run on hardware
//
#include <hip/hip_runtime.h>

#define BATCH 4
#define TLEN 4096
#define DMODEL 384
#define NQK 128                    // Yqk row: cols 0..63 = q, 64..127 = k
#define MTOT (BATCH * TLEN)        // 16384 tokens
#define NW 512                     // concat W rows: 64 q + 64 k + 384 v

typedef __attribute__((ext_vector_type(16))) _Float16 v16h;
typedef __attribute__((ext_vector_type(8)))  _Float16 v8h;
typedef __attribute__((ext_vector_type(4)))  _Float16 v4h;
typedef __attribute__((ext_vector_type(8)))  float    v8f;
typedef __attribute__((ext_vector_type(4)))  float    v4f;

__device__ __forceinline__ v16h combine16(v8h lo, v8h hi) {
    v16h r;
#pragma unroll
    for (int i = 0; i < 8; ++i) { r[i] = lo[i]; r[8 + i] = hi[i]; }
    return r;
}

// ---------------------------------------------------------------------------
// Kernel 0a/0b: one-time f32 -> f16 conversion of x and concat [Wq;Wk;Wv].
// Removes all v_cvt from the GEMM hot loops and halves L2 bytes per WMMA.
// ---------------------------------------------------------------------------
__global__ __launch_bounds__(256) void cvt_x_kernel(
    const float* __restrict__ x, _Float16* __restrict__ x16)
{
    const int i = (blockIdx.x * 256 + threadIdx.x) * 4;
    v4f v = *(const v4f*)(x + i);
    v4h h;
#pragma unroll
    for (int k = 0; k < 4; ++k) h[k] = (_Float16)v[k];
    *(v4h*)(x16 + i) = h;
}

__global__ __launch_bounds__(256) void cvt_w_kernel(
    const float* __restrict__ Wq, const float* __restrict__ Wk,
    const float* __restrict__ Wv, _Float16* __restrict__ Wc)
{
    const int i   = (blockIdx.x * 256 + threadIdx.x) * 4;
    const int row = i / DMODEL;
    const int col = i % DMODEL;
    const float* src;
    if (row < 64)       src = Wq + (size_t)row * DMODEL + col;
    else if (row < 128) src = Wk + (size_t)(row - 64) * DMODEL + col;
    else                src = Wv + (size_t)(row - 128) * DMODEL + col;
    v4f v = *(const v4f*)src;
    v4h h;
#pragma unroll
    for (int k = 0; k < 4; ++k) h[k] = (_Float16)v[k];
    *(v4h*)(Wc + i) = h;
}

// ---------------------------------------------------------------------------
// Kernel 1: Y = x16 @ Wc^T  (M=16384, N=512, K=384), all-f16 operands.
// One wave per 16x16 tile; 12 chained v_wmma_f32_16x16x32_f16, pure b128
// loads in the loop. Epilogue bounces C through padded LDS so Yqk (row major)
// and Vt (transposed, Vt[e][token]) are written with wide coalesced stores.
// ---------------------------------------------------------------------------
__global__ __launch_bounds__(256) void qkv_proj_kernel(
    const _Float16* __restrict__ x16, const _Float16* __restrict__ Wc,
    _Float16* __restrict__ Yqk, _Float16* __restrict__ Vt)
{
    __shared__ __align__(16) _Float16 ldsT[8][16][24];   // pad 24: conflict-free

    const int wave = threadIdx.x >> 5;
    const int lane = threadIdx.x & 31;
    const int lo   = lane & 15;
    const int hiA  = (lane >> 4) * 8;    // C row offset
    const int koB  = (lane >> 4) * 16;   // B K sub-chunk

    const int tile = blockIdx.x * 8 + wave;   // 1024 * 32 tiles
    const int m0 = (tile >> 5) * 16;
    const int n0 = (tile & 31) * 16;

    const _Float16* arow = x16 + (size_t)(m0 + lo) * DMODEL;
    const _Float16* brow = Wc  + (size_t)(n0 + lo) * DMODEL;

    v8f c;
#pragma unroll
    for (int i = 0; i < 8; ++i) c[i] = 0.0f;

#pragma unroll
    for (int d = 0; d < DMODEL; d += 32) {
        // A: halves 0..15 -> K=d+koB+h
        v16h a = combine16(*(const v8h*)(arow + d + koB),
                           *(const v8h*)(arow + d + koB + 8));
        // B: halves 0..15 -> K=d+koB+h
        v16h bb = combine16(*(const v8h*)(brow + d + koB),
                            *(const v8h*)(brow + d + koB + 8));
        c = __builtin_amdgcn_wmma_f32_16x16x32_f16(false, a, false, bb,
                                                   (short)0, c, false, false);
    }

    if (n0 < 128) {
        // LDS transpose C layout -> row-major, then 16B stores
#pragma unroll
        for (int j = 0; j < 8; ++j)
            ldsT[wave][j + hiA][lo] = (_Float16)c[j];
        asm volatile("s_wait_dscnt 0x0" ::: "memory");
        const int m   = lane >> 1;
        const int nch = (lane & 1) * 8;
        v8h row = *(const v8h*)&ldsT[wave][m][nch];
        *(v8h*)(Yqk + (size_t)(m0 + m) * NQK + n0 + nch) = row;
    } else {
        // transpose to Vt[e][token]: LDS as [e_local][token_local]
#pragma unroll
        for (int j = 0; j < 8; ++j)
            ldsT[wave][lo][j + hiA] = (_Float16)c[j];
        asm volatile("s_wait_dscnt 0x0" ::: "memory");
        const int tch = (lane >> 4) * 8;
        v8h row = *(const v8h*)&ldsT[wave][lo][tch];
        *(v8h*)(Vt + (size_t)(n0 - 128 + lo) * MTOT + m0 + tch) = row;
    }
}

// ---------------------------------------------------------------------------
// Kernel 2: causal flash attention, producer/consumer pipelined.
// Block = (batch, 16-row q tile), 4 waves. Wave 0 produces step i+1's
// P (f16, LDS, double-buffered) + per-row corr factors while all waves
// consume step i (PV WMMAs). One split barrier per 32-key step. Uneven V
// partition (3/7/7/7 e-tiles) balances the producer's extra softmax work.
// ---------------------------------------------------------------------------
__global__ __launch_bounds__(128) void attn_kernel(
    const _Float16* __restrict__ Yqk, const _Float16* __restrict__ Vt,
    float* __restrict__ out)
{
    __shared__ __align__(16) _Float16 ldsP[2][16][32];
    __shared__ float ldsCorr[2][16];
    __shared__ float ldsLinv[16];

    const int wave = threadIdx.x >> 5;
    const int lane = threadIdx.x & 31;
    const int lo   = lane & 15;
    const int hiA  = (lane >> 4) * 8;
    const int koB  = (lane >> 4) * 16;

    const int b      = blockIdx.x >> 8;
    const int qt     = blockIdx.x & 255;
    const int t0     = qt * 16;
    const int nsteps = (qt >> 1) + 1;

    const int myTiles = (wave == 0) ? 3 : 7;
    const int eStart  = (wave == 0) ? 0 : 48 + (wave - 1) * 112;

    const float scale = 0.125f;   // 1/sqrt(64)

    // producer-private state (wave 0)
    v16h aQ0{}, aQ1{};
    float m_run[8], l_run[8];
#pragma unroll
    for (int j = 0; j < 8; ++j) { m_run[j] = -1e30f; l_run[j] = 0.0f; }
    if (wave == 0) {
        const _Float16* qrow = Yqk + (size_t)(b * TLEN + t0 + lo) * NQK;
        aQ0 = combine16(*(const v8h*)(qrow + koB), *(const v8h*)(qrow + koB + 8));
        aQ1 = combine16(*(const v8h*)(qrow + 32 + koB), *(const v8h*)(qrow + 40 + koB));
    }

    v8f zc;
#pragma unroll
    for (int j = 0; j < 8; ++j) zc[j] = 0.0f;
    v8f acc[7];
#pragma unroll
    for (int e = 0; e < 7; ++e)
#pragma unroll
        for (int j = 0; j < 8; ++j) acc[e][j] = 0.0f;

    auto produce = [&](int step) {
        const int pb = step & 1;
        const int s0 = step * 32;
        const _Float16* k0row = Yqk + (size_t)(b * TLEN + s0 + lo) * NQK + 64;
        const _Float16* k1row = k0row + (size_t)16 * NQK;
        __builtin_prefetch(k0row + (size_t)32 * NQK, 0, 1);

        v16h bK00 = combine16(*(const v8h*)(k0row + koB),
                              *(const v8h*)(k0row + koB + 8));
        v16h bK01 = combine16(*(const v8h*)(k0row + 32 + koB),
                              *(const v8h*)(k0row + 40 + koB));
        v16h bK10 = combine16(*(const v8h*)(k1row + koB),
                              *(const v8h*)(k1row + koB + 8));
        v16h bK11 = combine16(*(const v8h*)(k1row + 32 + koB),
                              *(const v8h*)(k1row + 40 + koB));

        v8f s0v = __builtin_amdgcn_wmma_f32_16x16x32_f16(false, aQ0, false, bK00,
                                                         (short)0, zc, false, false);
        s0v = __builtin_amdgcn_wmma_f32_16x16x32_f16(false, aQ1, false, bK01,
                                                     (short)0, s0v, false, false);
        v8f s1v = __builtin_amdgcn_wmma_f32_16x16x32_f16(false, aQ0, false, bK10,
                                                         (short)0, zc, false, false);
        s1v = __builtin_amdgcn_wmma_f32_16x16x32_f16(false, aQ1, false, bK11,
                                                     (short)0, s1v, false, false);

#pragma unroll
        for (int j = 0; j < 8; ++j) {
            const int row = t0 + j + hiA;
            float v0 = ((s0 + lo) <= row)      ? s0v[j] * scale : -1e30f;
            float v1 = ((s0 + 16 + lo) <= row) ? s1v[j] * scale : -1e30f;

            float rm = fmaxf(v0, v1);
            rm = fmaxf(rm, __shfl_xor(rm, 1, 32));
            rm = fmaxf(rm, __shfl_xor(rm, 2, 32));
            rm = fmaxf(rm, __shfl_xor(rm, 4, 32));
            rm = fmaxf(rm, __shfl_xor(rm, 8, 32));

            const float nm   = fmaxf(m_run[j], rm);
            const float corr = __expf(m_run[j] - nm);
            m_run[j] = nm;

            const float e0 = __expf(v0 - nm);
            const float e1 = __expf(v1 - nm);
            float rs = e0 + e1;
            rs += __shfl_xor(rs, 1, 32);
            rs += __shfl_xor(rs, 2, 32);
            rs += __shfl_xor(rs, 4, 32);
            rs += __shfl_xor(rs, 8, 32);
            l_run[j] = l_run[j] * corr + rs;

            ldsP[pb][j + hiA][lo]      = (_Float16)e0;
            ldsP[pb][j + hiA][16 + lo] = (_Float16)e1;
            if (lo == 0) ldsCorr[pb][j + hiA] = corr;
        }
    };

    if (wave == 0) produce(0);
    __syncthreads();

    for (int i = 0; i < nsteps; ++i) {
        const int cb = i & 1;
        if (wave == 0 && (i + 1) < nsteps) produce(i + 1);

        // ---- consume step i ----
        const int s0 = i * 32;
        float cj[8];
#pragma unroll
        for (int j = 0; j < 8; ++j) cj[j] = ldsCorr[cb][j + hiA];

        const _Float16* prow = &ldsP[cb][lo][0];
        const v16h aP = combine16(*(const v8h*)(prow + koB),
                                  *(const v8h*)(prow + koB + 8));
#pragma unroll
        for (int et = 0; et < 7; ++et) {
            if (et < myTiles) {
#pragma unroll
                for (int j = 0; j < 8; ++j) acc[et][j] *= cj[j];
                const _Float16* vcol = Vt + (size_t)(eStart + et * 16 + lo) * MTOT
                                          + (size_t)(b * TLEN + s0) + koB;
                v16h bV = combine16(*(const v8h*)vcol, *(const v8h*)(vcol + 8));
                acc[et] = __builtin_amdgcn_wmma_f32_16x16x32_f16(false, aP, false, bV,
                                                                 (short)0, acc[et],
                                                                 false, false);
            }
        }
        __syncthreads();
    }

    if (wave == 0 && lo == 0) {
#pragma unroll
        for (int j = 0; j < 8; ++j) ldsLinv[j + hiA] = 1.0f / l_run[j];
    }
    __syncthreads();

    float linv[8];
#pragma unroll
    for (int j = 0; j < 8; ++j) linv[j] = ldsLinv[j + hiA];

#pragma unroll
    for (int et = 0; et < 7; ++et) {
        if (et < myTiles) {
            const int e = eStart + et * 16 + lo;
#pragma unroll
            for (int j = 0; j < 8; ++j) {
                const int row = t0 + j + hiA;
                out[(size_t)(b * TLEN + row) * DMODEL + e] = acc[et][j] * linv[j];
            }
        }
    }
}

extern "C" void kernel_launch(void* const* d_in, const int* in_sizes, int n_in,
                              void* d_out, int out_size, void* d_ws, size_t ws_size,
                              hipStream_t stream) {
    (void)in_sizes; (void)n_in; (void)out_size; (void)ws_size;
    const float* x  = (const float*)d_in[0];
    const float* Wk = (const float*)d_in[1];
    const float* Wq = (const float*)d_in[2];
    const float* Wv = (const float*)d_in[3];

    _Float16* x16 = (_Float16*)d_ws;                       // 16384 x 384 (12 MB)
    _Float16* Wc  = x16 + (size_t)MTOT * DMODEL;           // 512 x 384   (0.4 MB)
    _Float16* Yqk = Wc  + (size_t)NW * DMODEL;             // 16384 x 128 (4 MB)
    _Float16* Vt  = Yqk + (size_t)MTOT * NQK;              // 384 x 16384 (12 MB)
    float*    out = (float*)d_out;

    cvt_x_kernel<<<(MTOT * DMODEL) / (256 * 4), 256, 0, stream>>>(x, x16);
    cvt_w_kernel<<<(NW * DMODEL) / (256 * 4), 256, 0, stream>>>(Wq, Wk, Wv, Wc);
    qkv_proj_kernel<<<4096, 256, 0, stream>>>(x16, Wc, Yqk, Vt);
    attn_kernel<<<1024, 128, 0, stream>>>(Yqk, Vt, out);
}
